// Meg_InteractionBlockLoop_55130200211626
// MI455X (gfx1250) — matrix operand order repaired and checked by reading, not compiled, not run on hardware
//
#include <hip/hip_runtime.h>
#include <cstdint>
#include <cstddef>

// ---------------- model constants ----------------
#define HIDDEN  128
#define NGAUSS  50
#define NFILT   128
#define NSTATE  2
#define NCONV   2

// ---------------- WMMA types (CDNA5 / gfx1250, wave32) ----------------
typedef __attribute__((ext_vector_type(16))) __bf16 v16bf;
typedef __attribute__((ext_vector_type(8)))  float  v8f;

union Frag16 {
    v16bf v;
    unsigned short u[16];
    unsigned int   d[8];
    uint4          q[2];
};

// RNE f32->bf16 (weights, staged once per block)
__device__ __forceinline__ unsigned short f2bfu(float f) {
    unsigned int x = __builtin_bit_cast(unsigned int, f);
    unsigned int r = x + 0x7FFFu + ((x >> 16) & 1u);
    return (unsigned short)(r >> 16);
}

// pack two f32 into {bf16(f1), bf16(f0)} with one v_perm_b32 (truncation)
__device__ __forceinline__ unsigned int pk2bf(float f0, float f1) {
    return __builtin_amdgcn_perm(__builtin_bit_cast(unsigned int, f1),
                                 __builtin_bit_cast(unsigned int, f0),
                                 0x07060302u);
}

__device__ __forceinline__ unsigned short f2bfu_t(float f) {
    return (unsigned short)(__builtin_bit_cast(unsigned int, f) >> 16);
}

__device__ __forceinline__ float sspf(float x) {
    float sp = (x > 20.f) ? x : __logf(1.f + __expf(x));
    return sp - 0.69314718055994531f;
}

// Stage a [32*KK x 128] f32 weight into LDS in *fragment order*:
// u16 index ((kk*8 + j)*32 + lane)*16 + k  <=  W[(kk*32 + (lane>>4)*8 + k + (k>>3)*8)*128 + j*16 + (lane&15)]
// (rows >= validRows are zero-padded). Packed u32 stores.
template <int KK>
__device__ __forceinline__ void stage_swizzled(unsigned short* lds, const float* __restrict__ Wg,
                                               int validRows) {
    for (int i2 = threadIdx.x; i2 < KK * 2048; i2 += blockDim.x) {
        const int kp = i2 & 7;            // k pair
        const int ln = (i2 >> 3) & 31;
        const int j  = (i2 >> 8) & 7;
        const int kk = i2 >> 11;
        const int lo_ = ln & 15, hi_ = ln >> 4;
        const int r0 = kk * 32 + hi_ * 8 + 2 * kp + ((kp >> 2) << 3);
        const int c  = j * 16 + lo_;
        unsigned int w0 = (r0     < validRows) ? f2bfu(Wg[(r0)     * 128 + c]) : 0u;
        unsigned int w1 = (r0 + 1 < validRows) ? f2bfu(Wg[(r0 + 1) * 128 + c]) : 0u;
        ((unsigned int*)lds)[i2] = w0 | (w1 << 16);
    }
}

enum { MODE_LIN1 = 0, MODE_PLAIN = 1, MODE_SSP = 2, MODE_RES = 3, MODE_RELU = 4 };

// =====================================================================
// Node GEMM: Y[N,128] = epilogue( X[N,128] @ W[128,128] + b )
// One wave computes a 32x128 block (two 16-row subtiles sharing each
// B fragment): 64 wmma per 64 ds_load_b128.
// =====================================================================
template <int MODE>
__global__ __launch_bounds__(256)
void meg_node_gemm(const float* __restrict__ X, const float* __restrict__ Wg,
                   const float* __restrict__ bias, float* __restrict__ Y,
                   const float* __restrict__ resid,        // MODE_RES
                   const int*   __restrict__ batch,        // MODE_LIN1
                   const float* __restrict__ sa,           // MODE_LIN1: [G,2]
                   const float* __restrict__ w01,          // MODE_LIN1: 2 state rows
                   int nrows)
{
    __shared__ __align__(16) unsigned short Wlds[128 * 128];   // 32 KB, fragment order
    stage_swizzled<4>(Wlds, Wg, 128);
    __syncthreads();

    const int wave = threadIdx.x >> 5;
    const int lane = threadIdx.x & 31;
    const int lo = lane & 15;
    const int hi = lane >> 4;
    const int nt32 = (nrows + 31) >> 5;

    for (int t = blockIdx.x * 8 + wave; t < nt32; t += gridDim.x * 8) {
        const int row0 = t * 32;
        v8f acc[2][8];
        v8f zero = {};
#pragma unroll
        for (int s = 0; s < 2; ++s)
#pragma unroll
            for (int j = 0; j < 8; ++j) acc[s][j] = zero;

        int ar0 = row0 + lo;       if (ar0 >= nrows) ar0 = nrows - 1;
        int ar1 = row0 + 16 + lo;  if (ar1 >= nrows) ar1 = nrows - 1;
        const float* __restrict__ x0 = X + (size_t)ar0 * 128;
        const float* __restrict__ x1 = X + (size_t)ar1 * 128;

#pragma unroll
        for (int kk = 0; kk < 4; ++kk) {
            Frag16 a0, a1;
            const float* p0 = x0 + kk * 32 + 8 * hi;
            const float* p1 = x1 + kk * 32 + 8 * hi;
#pragma unroll
            for (int p = 0; p < 4; ++p) {
                a0.d[p]     = pk2bf(p0[2 * p],      p0[2 * p + 1]);
                a0.d[4 + p] = pk2bf(p0[16 + 2 * p], p0[16 + 2 * p + 1]);
                a1.d[p]     = pk2bf(p1[2 * p],      p1[2 * p + 1]);
                a1.d[4 + p] = pk2bf(p1[16 + 2 * p], p1[16 + 2 * p + 1]);
            }
#pragma unroll
            for (int j = 0; j < 8; ++j) {
                Frag16 b;
                const uint4* bp = (const uint4*)(Wlds + (((kk * 8 + j) * 32 + lane) << 4));
                b.q[0] = bp[0];
                b.q[1] = bp[1];
                acc[0][j] = __builtin_amdgcn_wmma_f32_16x16x32_bf16(
                    false, a0.v, false, b.v, (short)0, acc[0][j], false, false);
                acc[1][j] = __builtin_amdgcn_wmma_f32_16x16x32_bf16(
                    false, a1.v, false, b.v, (short)0, acc[1][j], false, false);
            }
        }

        // ----- epilogue (C/D layout: lane holds col n = j*16+lo, rows r+8*hi) -----
#pragma unroll
        for (int s = 0; s < 2; ++s) {
#pragma unroll
            for (int j = 0; j < 8; ++j) {
                const int n = j * 16 + lo;
                float bn = 0.f;
                if (MODE != MODE_PLAIN) bn = bias[n];
                float w0 = 0.f, w1 = 0.f;
                if (MODE == MODE_LIN1) { w0 = w01[n]; w1 = w01[128 + n]; }
#pragma unroll
                for (int r = 0; r < 8; ++r) {
                    const int row = row0 + s * 16 + r + 8 * hi;
                    if (row < nrows) {
                        float v = acc[s][j][r] + bn;
                        if (MODE == MODE_LIN1) {
                            const int g = batch[row];
                            v += sa[2 * g] * w0 + sa[2 * g + 1] * w1;
                        }
                        if (MODE == MODE_SSP)  v = sspf(v);
                        if (MODE == MODE_RELU) v = fmaxf(v, 0.f);
                        if (MODE == MODE_RES)  v += resid[(size_t)row * 128 + n];
                        Y[(size_t)row * 128 + n] = v;
                    }
                }
            }
        }
    }
}

// =====================================================================
// Fused edge kernel (per 16-edge tile / wave):
//   T = ssp(edge_attr @ mlp1_w + b1)   [16x128], K=50 padded to 64
//   W = (T @ mlp2_w + b2) * cutoff(e)  [16x128], never hits HBM
//   atomicAdd(agg[dst], xm[src] * W)
// =====================================================================
__global__ __launch_bounds__(256)
void meg_edge_kernel(const float* __restrict__ edge_attr,
                     const float* __restrict__ edge_weight,
                     const int*   __restrict__ edge_index,   // [2,E]
                     const float* __restrict__ mlp1_w,       // [50,128]
                     const float* __restrict__ mlp1_b,
                     const float* __restrict__ mlp2_w,       // [128,128]
                     const float* __restrict__ mlp2_b,
                     const float* __restrict__ xm,           // [N,128]
                     float*       __restrict__ agg,          // [N,128]
                     int E)
{
    __shared__ __align__(16) unsigned short W1[64 * 128];      // 16 KB, fragment order
    __shared__ __align__(16) unsigned short W2[128 * 128];     // 32 KB, fragment order
    __shared__ __align__(16) unsigned short Tscr[8][16 * 32];  // 8 KB per-wave relayout

    stage_swizzled<2>(W1, mlp1_w, NGAUSS);
    stage_swizzled<4>(W2, mlp2_w, 128);
    __syncthreads();

    const int wave = threadIdx.x >> 5;
    const int lane = threadIdx.x & 31;
    const int lo = lane & 15;
    const int hi = lane >> 4;
    const int ntiles = (E + 15) >> 4;

    for (int tile = blockIdx.x * 8 + wave; tile < ntiles; tile += gridDim.x * 8) {
        const int e0 = tile * 16;

        {   // prefetch next tile's edge_attr rows into GL2
            long pe = (long)(e0 + 16 * gridDim.x * 8) + lo;
            if (pe < E) __builtin_prefetch(edge_attr + pe * NGAUSS, 0, 1);
        }

        // ---- per-row metadata (this lane's rows m = r + 8*hi) ----
        int srcr[8], dstr[8];
        float cf[8];
#pragma unroll
        for (int r = 0; r < 8; ++r) {
            const int e = e0 + r + 8 * hi;
            const int ec = (e < E) ? e : 0;
            srcr[r] = edge_index[ec];
            dstr[r] = edge_index[E + ec];
            const float w = edge_weight[ec];
            cf[r] = 0.5f * (__cosf(w * 0.314159265358979f) + 1.f);
        }

        // ---- GEMM 1: edge_attr[16x50] @ W1 (K padded to 64, 2 steps) ----
        v8f acc1[8];
        v8f zero = {};
#pragma unroll
        for (int j = 0; j < 8; ++j) acc1[j] = zero;

        int ae = e0 + lo;
        if (ae >= E) ae = E - 1;
        const float* __restrict__ arow = edge_attr + (size_t)ae * NGAUSS;

#pragma unroll
        for (int kk = 0; kk < 2; ++kk) {
            Frag16 a;
            const float* p0 = arow + kk * 32 + 8 * hi;   // chunk0: cols <= 47, always valid
#pragma unroll
            for (int p = 0; p < 4; ++p) a.d[p] = pk2bf(p0[2 * p], p0[2 * p + 1]);
            if (kk == 0) {                                // chunk1: cols 16+8*hi+.. <= 31, valid
#pragma unroll
                for (int p = 0; p < 4; ++p) a.d[4 + p] = pk2bf(p0[16 + 2 * p], p0[16 + 2 * p + 1]);
            } else {                                      // chunk1: cols 48..63 -> only 48,49 valid
                const float t0 = arow[48], t1 = arow[49]; // safe addresses for all lanes
                a.d[4] = (hi == 0) ? pk2bf(t0, t1) : 0u;
                a.d[5] = 0u; a.d[6] = 0u; a.d[7] = 0u;
            }
#pragma unroll
            for (int j = 0; j < 8; ++j) {
                Frag16 b;
                const uint4* bp = (const uint4*)(W1 + (((kk * 8 + j) * 32 + lane) << 4));
                b.q[0] = bp[0];
                b.q[1] = bp[1];
                acc1[j] = __builtin_amdgcn_wmma_f32_16x16x32_bf16(
                    false, a.v, false, b.v, (short)0, acc1[j], false, false);
            }
        }

        // ---- bias + ssp, C->A relayout via per-wave LDS, GEMM 2 (K=128) ----
        v8f acc2[8];
#pragma unroll
        for (int j = 0; j < 8; ++j) acc2[j] = zero;

#pragma unroll
        for (int kk2 = 0; kk2 < 4; ++kk2) {
#pragma unroll
            for (int jj = 0; jj < 2; ++jj) {
                const int j = 2 * kk2 + jj;
                const float bn = mlp1_b[j * 16 + lo];
#pragma unroll
                for (int r = 0; r < 8; ++r) {
                    const float tv = sspf(acc1[j][r] + bn);
                    Tscr[wave][(r + 8 * hi) * 32 + jj * 16 + lo] = f2bfu_t(tv);
                }
            }
            // same-wave LDS ops are in-order; fence before fragment reload
            asm volatile("s_wait_dscnt 0" ::: "memory");

            Frag16 a2;
            const unsigned short* sp = &Tscr[wave][lo * 32 + 8 * hi];
            a2.q[0] = *(const uint4*)(sp);
            a2.q[1] = *(const uint4*)(sp + 16);
#pragma unroll
            for (int j = 0; j < 8; ++j) {
                Frag16 b;
                const uint4* bp = (const uint4*)(W2 + (((kk2 * 8 + j) * 32 + lane) << 4));
                b.q[0] = bp[0];
                b.q[1] = bp[1];
                acc2[j] = __builtin_amdgcn_wmma_f32_16x16x32_bf16(
                    false, a2.v, false, b.v, (short)0, acc2[j], false, false);
            }
        }

        // ---- epilogue: W = (acc2+b2)*cutoff ; gather xm[src] ; scatter-add ----
#pragma unroll
        for (int j = 0; j < 8; ++j) {
            const int n = j * 16 + lo;
            const float b2 = mlp2_b[n];
#pragma unroll
            for (int r = 0; r < 8; ++r) {
                const int e = e0 + r + 8 * hi;
                if (e < E) {
                    const float wv = (acc2[j][r] + b2) * cf[r];
                    const float xv = xm[(size_t)srcr[r] * 128 + n];
                    atomicAdd(&agg[(size_t)dstr[r] * 128 + n], xv * wv);
                }
            }
        }
    }
}

// ---------------- small utility kernels ----------------
__global__ void meg_zero(float* p, long n) {
    long i = (long)blockIdx.x * blockDim.x + threadIdx.x;
    long stride = (long)gridDim.x * blockDim.x;
    for (; i < n; i += stride) p[i] = 0.f;
}

__global__ void meg_count(const int* __restrict__ batch, float* counts, int n) {
    int i = blockIdx.x * blockDim.x + threadIdx.x;
    if (i < n) atomicAdd(&counts[batch[i]], 1.f);
}

__global__ __launch_bounds__(256)
void meg_rowsum(const float* __restrict__ h, const int* __restrict__ batch,
                float* ssum, int n) {
    const int wave = threadIdx.x >> 5;
    const int lane = threadIdx.x & 31;
    for (int node = blockIdx.x * 8 + wave; node < n; node += gridDim.x * 8) {
        const float* r = h + (size_t)node * 128;
        float v = r[lane] + r[lane + 32] + r[lane + 64] + r[lane + 96];
#pragma unroll
        for (int off = 16; off > 0; off >>= 1) v += __shfl_xor(v, off, 32);
        if (lane == 0) atomicAdd(&ssum[batch[node]], v);
    }
}

__global__ void meg_state(const float* ssum, const float* counts, float* sab, int G) {
    int g = blockIdx.x * blockDim.x + threadIdx.x;
    if (g < G) {
        float s = ssum[g] / fmaxf(counts[g], 1.f);
        sab[2 * g] = s;
        sab[2 * g + 1] = s;
    }
}

// =====================================================================
extern "C" void kernel_launch(void* const* d_in, const int* in_sizes, int n_in,
                              void* d_out, int out_size, void* d_ws, size_t ws_size,
                              hipStream_t stream)
{
    const float* h_in       = (const float*)d_in[0];
    const int*   edge_index = (const int*)  d_in[1];
    const float* edge_w     = (const float*)d_in[2];
    const float* edge_attr  = (const float*)d_in[3];
    const float* state_attr = (const float*)d_in[4];
    const int*   batch      = (const int*)  d_in[5];
    const float* lin1_w     = (const float*)d_in[6];
    const float* lin1_b     = (const float*)d_in[7];
    const float* mlp1_w     = (const float*)d_in[8];
    const float* mlp1_b     = (const float*)d_in[9];
    const float* mlp2_w     = (const float*)d_in[10];
    const float* mlp2_b     = (const float*)d_in[11];
    const float* conv1_w    = (const float*)d_in[12];
    const float* conv2_w    = (const float*)d_in[13];
    const float* conv2_b    = (const float*)d_in[14];
    const float* blin_w     = (const float*)d_in[15];
    const float* blin_b     = (const float*)d_in[16];
    const float* out_w      = (const float*)d_in[17];
    const float* out_b      = (const float*)d_in[18];

    const int N = in_sizes[0] / HIDDEN;
    const int E = in_sizes[2];
    const int G = in_sizes[4] / NSTATE;
    const size_t nf = (size_t)N * HIDDEN;

    float* B0     = (float*)d_ws;        // h1 (layer 0) -> h after layer 0
    float* B1     = B0 + nf;             // xm / ssp(conv2) scratch
    float* B2     = B1 + nf;             // agg
    float* B3     = B2 + nf;             // h1 (layer 1) -> final h
    float* counts = B3 + nf;             // [G]
    float* ssum   = counts + G;          // [G]
    float* sab    = ssum + G;            // [G,2]

    const int nt32    = (N + 31) / 32;
    const int gblocks = (nt32 + 7) / 8;
    const int eblocks = 1024;
    const int zblocks = 2048;

    meg_zero<<<4, 256, 0, stream>>>(counts, (long)G * 2);   // counts + ssum
    meg_count<<<(N + 255) / 256, 256, 0, stream>>>(batch, counts, N);

    const float* hPrev = h_in;
    const float* saCur = state_attr;
    float* h1 = B0;

    for (int l = 0; l < NCONV; ++l) {
        const float* l1w  = lin1_w + (size_t)l * (HIDDEN + NSTATE) * HIDDEN;
        const float* l1wg = l1w + NSTATE * HIDDEN;
        const float* l1b  = lin1_b + (size_t)l * HIDDEN;
        const float* m1w  = mlp1_w + (size_t)l * NGAUSS * NFILT;
        const float* m1b  = mlp1_b + (size_t)l * NFILT;
        const float* m2w  = mlp2_w + (size_t)l * NFILT * NFILT;
        const float* m2b  = mlp2_b + (size_t)l * NFILT;
        const float* c1w  = conv1_w + (size_t)l * HIDDEN * NFILT;
        const float* c2w  = conv2_w + (size_t)l * NFILT * HIDDEN;
        const float* c2b  = conv2_b + (size_t)l * HIDDEN;
        const float* blw  = blin_w + (size_t)l * HIDDEN * HIDDEN;
        const float* blb  = blin_b + (size_t)l * HIDDEN;

        h1 = (l == 0) ? B0 : B3;

        meg_node_gemm<MODE_LIN1><<<gblocks, 256, 0, stream>>>(
            hPrev, l1wg, l1b, h1, nullptr, batch, saCur, l1w, N);
        meg_node_gemm<MODE_PLAIN><<<gblocks, 256, 0, stream>>>(
            h1, c1w, nullptr, B1, nullptr, nullptr, nullptr, nullptr, N);
        meg_zero<<<zblocks, 256, 0, stream>>>(B2, (long)nf);
        meg_edge_kernel<<<eblocks, 256, 0, stream>>>(
            edge_attr, edge_w, edge_index, m1w, m1b, m2w, m2b, B1, B2, E);
        meg_node_gemm<MODE_SSP><<<gblocks, 256, 0, stream>>>(
            B2, c2w, c2b, B1, nullptr, nullptr, nullptr, nullptr, N);
        meg_node_gemm<MODE_RES><<<gblocks, 256, 0, stream>>>(
            B1, blw, blb, h1, h1, nullptr, nullptr, nullptr, N);

        if (l + 1 < NCONV) {
            meg_zero<<<4, 256, 0, stream>>>(ssum, (long)G);
            meg_rowsum<<<512, 256, 0, stream>>>(h1, batch, ssum, N);
            meg_state<<<(G + 255) / 256, 256, 0, stream>>>(ssum, counts, sab, G);
            saCur = sab;
            hPrev = h1;
        }
    }

    meg_node_gemm<MODE_RELU><<<gblocks, 256, 0, stream>>>(
        h1, out_w, out_b, (float*)d_out, nullptr, nullptr, nullptr, nullptr, N);
}
